// LSTM_60687887893313
// MI455X (gfx1250) — matrix operand order repaired and checked by reading, not compiled, not run on hardware
//
#include <hip/hip_runtime.h>
#include <hip/hip_bf16.h>

typedef __attribute__((ext_vector_type(16))) __bf16 v16bf;
typedef __attribute__((ext_vector_type(8)))  __bf16 v8bf;
typedef __attribute__((ext_vector_type(2)))  __bf16 v2bf;
typedef __attribute__((ext_vector_type(8)))  float  v8f;

#define HID  32
#define NL   8
#define DIN  64
#define FEAT 4096
#define TT   300
#define BB   64
#define BT   (BB * TT)   // 19200

#define CAT16(lo, hi) __builtin_shufflevector((lo), (hi), 0,1,2,3,4,5,6,7,8,9,10,11,12,13,14,15)

#define WMMA_BF16(a, b, c) __builtin_amdgcn_wmma_f32_16x16x32_bf16( \
    false, (a), false, (b), (short)0, (c), false, false)

// ---------------------------------------------------------------------------
// Kernel 0: zero the inter-layer handshake flags
// ---------------------------------------------------------------------------
__global__ void zero_flags_kernel(int* flags, int n) {
    int i = blockIdx.x * blockDim.x + threadIdx.x;
    if (i < n) flags[i] = 0;
}

// ---------------------------------------------------------------------------
// Kernel 1: fc0 = Linear(4096->64) + ReLU + Linear(64->64), output bf16
//  grid 150 x 256 threads (8 waves). Each block: 128 rows of [19200 x 64].
//  Software-pipelined K loop with double-buffered LDS (1 barrier / K-iter,
//  global loads for chunk k+1 in flight during WMMA of chunk k).
//  Output tile leaves via async LDS->global DMA (ASYNCcnt).
// ---------------------------------------------------------------------------
__global__ __launch_bounds__(256)
void fc0_kernel(const float* __restrict__ x, const float* __restrict__ W0,
                const float* __restrict__ b0, const float* __restrict__ W1,
                const float* __restrict__ b1, __bf16* __restrict__ hbuf)
{
    __shared__ alignas(32) __bf16 Xs[2][128][32];   // x chunk [row][k], double buffered
    __shared__ alignas(32) __bf16 WsT[2][64][32];   // W0 chunk [n][k], double buffered
    __shared__ alignas(32) __bf16 W1T[64][64];      // W1 [n][k]
    __shared__ alignas(32) __bf16 Y1[128][64];      // relu(stage1) / stage2 result

    const int tid  = threadIdx.x;
    const int lane = tid & 31;
    const int wave = tid >> 5;
    const int m0   = wave * 16;
    const int rowBase = blockIdx.x * 128;

    const int arow = lane & 15;          // A-frag row within tile
    const int kb   = (lane >> 4) * 16;   // A-frag K base (lane half)
    const int kbB  = (lane >> 4) * 16;   // B-frag K base (lane half)
    const int ncol = lane & 15;          // B/C/D fragment column

    // W1^T -> LDS once (packed pair stores)
    for (int i = tid; i < 64 * 32; i += 256) {
        int n = i >> 5, k2 = (i & 31) * 2;
        v2bf p;
        p[0] = (__bf16)W1[n * 64 + k2];
        p[1] = (__bf16)W1[n * 64 + k2 + 1];
        *(v2bf*)&W1T[n][k2] = p;
    }

    // stage-1 accumulators, init with b0 broadcast
    v8f acc[4];
    #pragma unroll
    for (int nt = 0; nt < 4; ++nt) {
        float bv = b0[nt * 16 + ncol];
        #pragma unroll
        for (int r = 0; r < 8; ++r) acc[nt][r] = bv;
    }

    // --- pipelined K loop ---
    const int xr  = tid >> 1;             // staging row
    const int xk0 = (tid & 1) * 16;       // staging k base
    const int wn  = tid >> 2;             // W0 staging n
    const int wk0 = (tid & 3) * 8;        // W0 staging k base
    const float* xsrc = x  + (size_t)(rowBase + xr) * FEAT + xk0;
    const float* wsrc = W0 + (size_t)wn * FEAT + wk0;

    float4 c0 = ((const float4*)xsrc)[0];
    float4 c1 = ((const float4*)xsrc)[1];
    float4 c2 = ((const float4*)xsrc)[2];
    float4 c3 = ((const float4*)xsrc)[3];
    float4 w0r = ((const float4*)wsrc)[0];
    float4 w1r = ((const float4*)wsrc)[1];

    int buf = 0;
    for (int kc = 0; kc < FEAT; kc += 32, buf ^= 1) {
        // pack current chunk -> LDS[buf]
        v8bf p0, p1, pw;
        p0[0]=(__bf16)c0.x; p0[1]=(__bf16)c0.y; p0[2]=(__bf16)c0.z; p0[3]=(__bf16)c0.w;
        p0[4]=(__bf16)c1.x; p0[5]=(__bf16)c1.y; p0[6]=(__bf16)c1.z; p0[7]=(__bf16)c1.w;
        p1[0]=(__bf16)c2.x; p1[1]=(__bf16)c2.y; p1[2]=(__bf16)c2.z; p1[3]=(__bf16)c2.w;
        p1[4]=(__bf16)c3.x; p1[5]=(__bf16)c3.y; p1[6]=(__bf16)c3.z; p1[7]=(__bf16)c3.w;
        pw[0]=(__bf16)w0r.x; pw[1]=(__bf16)w0r.y; pw[2]=(__bf16)w0r.z; pw[3]=(__bf16)w0r.w;
        pw[4]=(__bf16)w1r.x; pw[5]=(__bf16)w1r.y; pw[6]=(__bf16)w1r.z; pw[7]=(__bf16)w1r.w;
        *(v8bf*)&Xs[buf][xr][xk0]     = p0;
        *(v8bf*)&Xs[buf][xr][xk0 + 8] = p1;
        *(v8bf*)&WsT[buf][wn][wk0]    = pw;

        // issue next chunk's global loads (waited only next iteration)
        if (kc + 32 < FEAT) {
            const float* nx = xsrc + kc + 32;
            c0 = ((const float4*)nx)[0];
            c1 = ((const float4*)nx)[1];
            c2 = ((const float4*)nx)[2];
            c3 = ((const float4*)nx)[3];
            const float* nw = wsrc + kc + 32;
            w0r = ((const float4*)nw)[0];
            w1r = ((const float4*)nw)[1];
            if (kc + 64 < FEAT) __builtin_prefetch(xsrc + kc + 64, 0, 1);
        }
        __syncthreads();

        // fragments + WMMA on chunk kc
        v8bf alo = *(const v8bf*)&Xs[buf][m0 + arow][kb];
        v8bf ahi = *(const v8bf*)&Xs[buf][m0 + arow][8 + kb];
        v16bf a = CAT16(alo, ahi);
        #pragma unroll
        for (int nt = 0; nt < 4; ++nt) {
            v16bf bf = *(const v16bf*)&WsT[buf][nt * 16 + ncol][kbB];
            acc[nt] = WMMA_BF16(a, bf, acc[nt]);
        }
    }

    // ReLU + stash stage-1 result as bf16 in LDS
    #pragma unroll
    for (int nt = 0; nt < 4; ++nt)
        #pragma unroll
        for (int r = 0; r < 8; ++r) {
            int m = m0 + r + 8 * (lane >> 4);
            int n = nt * 16 + ncol;
            float v = acc[nt][r];
            Y1[m][n] = (__bf16)(v > 0.f ? v : 0.f);
        }
    __syncthreads();

    // stage 2: [128,64] x [64,64], K=64 -> 2 bf16 WMMA k-steps
    v8f acc2[4];
    #pragma unroll
    for (int nt = 0; nt < 4; ++nt) {
        float bv = b1[nt * 16 + ncol];
        #pragma unroll
        for (int r = 0; r < 8; ++r) acc2[nt][r] = bv;
    }
    #pragma unroll
    for (int kk = 0; kk < 2; ++kk) {
        v8bf alo = *(const v8bf*)&Y1[m0 + arow][kk * 32 + kb];
        v8bf ahi = *(const v8bf*)&Y1[m0 + arow][kk * 32 + 8 + kb];
        v16bf a = CAT16(alo, ahi);
        #pragma unroll
        for (int nt = 0; nt < 4; ++nt) {
            v16bf bf = *(const v16bf*)&W1T[nt * 16 + ncol][kk * 32 + kbB];
            acc2[nt] = WMMA_BF16(a, bf, acc2[nt]);
        }
    }
    __syncthreads();  // all Y1 reads done
    #pragma unroll
    for (int nt = 0; nt < 4; ++nt)
        #pragma unroll
        for (int r = 0; r < 8; ++r) {
            int m = m0 + r + 8 * (lane >> 4);
            int n = nt * 16 + ncol;
            Y1[m][n] = (__bf16)acc2[nt][r];
        }
    __syncthreads();

    // async DMA the 128x64 bf16 tile LDS -> global (ASYNCcnt path)
    for (int i = tid; i < 1024; i += 256) {
        int row = i >> 3, k0 = (i & 7) * 8;
        const __bf16* gp = hbuf + (size_t)(rowBase + row) * DIN + k0;
        unsigned loff = (unsigned)(uintptr_t)&Y1[row][k0];  // low 32b = LDS addr
        asm volatile("global_store_async_from_lds_b128 %0, %1, off"
                     :: "v"(gp), "v"(loff) : "memory");
    }
    asm volatile("s_wait_asynccnt 0x0" ::: "memory");
}

// ---------------------------------------------------------------------------
// Kernel 2: persistent pipelined 8-layer LSTM (wavefront over (layer,t)).
//  grid = 32 blocks: (layer l = bid>>2, batch-group bg = bid&3), 16 batches.
//  Critical path per step: 1 h-WMMA -> act -> barrier -> update -> barrier.
//  The x*Wx gate contribution for step t+1 (loads + 1-2 WMMAs) is computed
//  off the critical path. B fragments live in VGPRs for all 300 steps.
// ---------------------------------------------------------------------------
__global__ __launch_bounds__(256)
void lstm_kernel(const __bf16* __restrict__ hbuf,
                 const float* __restrict__ Wih0, const float* __restrict__ Wih,
                 const float* __restrict__ Whh,  const float* __restrict__ bih,
                 const float* __restrict__ bhh,
                 __bf16* __restrict__ hlay, int* __restrict__ flags)
{
    __shared__ alignas(32) __bf16 WtT[128][96];  // [gate-col][k], k = [x | h]
    __shared__ float  biasS[128];
    __shared__ alignas(32) __bf16 hb[16][32];    // h_{t-1} (bf16, A-frag source)
    __shared__ float  cs[16][32];                // cell state
    __shared__ float  act[4][16][32];            // activated gates i,f,g,o

    const int tid  = threadIdx.x;
    const int lane = tid & 31;
    const int wave = tid >> 5;
    const int l    = blockIdx.x >> 2;
    const int bg   = blockIdx.x & 3;
    const int b0r  = bg * 16;
    const int in   = (l == 0) ? DIN : HID;   // block-uniform
    const int K    = in + HID;

    const float* WihL = (l == 0) ? Wih0 : (Wih + (size_t)(l - 1) * 128 * HID);
    const float* WhhL = Whh + (size_t)l * 128 * HID;

    for (int i = tid; i < 128 * 96; i += 256) {
        int n = i / 96, k = i % 96;
        float w = (k < in) ? WihL[n * in + k]
                           : ((k < K) ? WhhL[n * HID + (k - in)] : 0.f);
        WtT[n][k] = (__bf16)w;
    }
    for (int i = tid; i < 128; i += 256)
        biasS[i] = bih[l * 128 + i] + bhh[l * 128 + i];
    for (int i = tid; i < 256; i += 256) {
        int m = i >> 4, n2 = (i & 15) * 2;
        cs[m][n2] = 0.f; cs[m][n2 + 1] = 0.f;
        v2bf z; z[0] = (__bf16)0.f; z[1] = (__bf16)0.f;
        *(v2bf*)&hb[m][n2] = z;
    }
    __syncthreads();

    // loop-invariant B fragments: 16 contiguous k at fixed n -> 32B LDS loads
    const int n0   = wave * 16;
    const int ncol = lane & 15;
    const int kbB  = (lane >> 4) * 16;
    const v16bf bfr0 = *(const v16bf*)&WtT[n0 + ncol][kbB];
    const v16bf bfr1 = *(const v16bf*)&WtT[n0 + ncol][32 + kbB];
    const v16bf bfr2 = *(const v16bf*)&WtT[n0 + ncol][64 + kbB];
    const v16bf bfrH = (in == DIN) ? bfr2 : bfr1;   // h-part weights

    const __bf16* inbuf  = (l == 0) ? hbuf : (hlay + (size_t)(l - 1) * BT * HID);
    __bf16*       outbuf = hlay + (size_t)l * BT * HID;
    int* fin  = (l > 0) ? (flags + ((l - 1) * 4 + bg) * TT) : (int*)0;
    int* fout = flags + (l * 4 + bg) * TT;
    const int  arow = lane & 15;
    const int  kb   = (lane >> 4) * 16;
    const float bv  = biasS[n0 + ncol];
    const int  gate = wave >> 1;
    const int  sub  = (wave & 1) * 16 + ncol;

    // ---- prologue: gx = bias + x_0 * Wx ----
    if (l > 0)
        while (__hip_atomic_load(fin, __ATOMIC_ACQUIRE,
                                 __HIP_MEMORY_SCOPE_AGENT) == 0)
            __builtin_amdgcn_s_sleep(1);
    v8f gx;
    #pragma unroll
    for (int r = 0; r < 8; ++r) gx[r] = bv;
    {
        const __bf16* xrow = inbuf + ((size_t)(b0r + arow) * TT) * in;
        v8bf lo = *(const v8bf*)(xrow + kb);
        v8bf hi = *(const v8bf*)(xrow + 8 + kb);
        gx = WMMA_BF16(CAT16(lo, hi), bfr0, gx);
        if (in == DIN) {
            lo = *(const v8bf*)(xrow + 32 + kb);
            hi = *(const v8bf*)(xrow + 40 + kb);
            gx = WMMA_BF16(CAT16(lo, hi), bfr1, gx);
        }
    }

    for (int t = 0; t < TT; ++t) {
        // --- critical path: g = gx + h_{t-1} * Wh ---
        v8bf hlo = *(const v8bf*)&hb[arow][kb];
        v8bf hhi = *(const v8bf*)&hb[arow][8 + kb];
        v8f g = WMMA_BF16(CAT16(hlo, hhi), bfrH, gx);

        // --- off critical path: gx for t+1 ---
        if (t + 1 < TT) {
            if (l > 0)
                while (__hip_atomic_load(fin + t + 1, __ATOMIC_ACQUIRE,
                                         __HIP_MEMORY_SCOPE_AGENT) == 0)
                    __builtin_amdgcn_s_sleep(1);
            #pragma unroll
            for (int r = 0; r < 8; ++r) gx[r] = bv;
            const __bf16* xrow = inbuf + ((size_t)(b0r + arow) * TT + t + 1) * in;
            v8bf lo = *(const v8bf*)(xrow + kb);
            v8bf hi = *(const v8bf*)(xrow + 8 + kb);
            gx = WMMA_BF16(CAT16(lo, hi), bfr0, gx);
            if (in == DIN) {
                lo = *(const v8bf*)(xrow + 32 + kb);
                hi = *(const v8bf*)(xrow + 40 + kb);
                gx = WMMA_BF16(CAT16(lo, hi), bfr1, gx);
            }
        }

        // gate nonlinearities (wave pair -> one gate: i,f,g,o)
        #pragma unroll
        for (int r = 0; r < 8; ++r) {
            int m = r + 8 * (lane >> 4);
            float v = g[r];
            float av = (gate == 2) ? tanhf(v) : (1.f / (1.f + __expf(-v)));
            act[gate][m][sub] = av;
        }
        __syncthreads();

        // cell / hidden update: one (m, n-pair) per thread, packed stores
        {
            int m = tid >> 4, n2 = (tid & 15) * 2;
            float c0 = act[1][m][n2]     * cs[m][n2]     + act[0][m][n2]     * act[2][m][n2];
            float c1 = act[1][m][n2 + 1] * cs[m][n2 + 1] + act[0][m][n2 + 1] * act[2][m][n2 + 1];
            float h0 = act[3][m][n2]     * tanhf(c0);
            float h1 = act[3][m][n2 + 1] * tanhf(c1);
            cs[m][n2] = c0; cs[m][n2 + 1] = c1;
            v2bf hv; hv[0] = (__bf16)h0; hv[1] = (__bf16)h1;
            *(v2bf*)&hb[m][n2] = hv;
            *(v2bf*)(outbuf + ((size_t)(b0r + m) * TT + t) * HID + n2) = hv;
        }
        asm volatile("s_wait_storecnt 0x0" ::: "memory");  // h_t globally visible
        __syncthreads();
        if (l < NL - 1 && tid == 0)
            __hip_atomic_store(fout + t, 1, __ATOMIC_RELEASE,
                               __HIP_MEMORY_SCOPE_AGENT);
    }
}

// ---------------------------------------------------------------------------
// Kernel 3: head  y[b] = sum_t Wfc2[t]*(bfc1 + h8[b,t,:].Wfc1) + bfc2
// ---------------------------------------------------------------------------
__global__ __launch_bounds__(128)
void head_kernel(const __bf16* __restrict__ h8, const float* __restrict__ Wfc1,
                 const float* __restrict__ bfc1, const float* __restrict__ Wfc2,
                 const float* __restrict__ bfc2, float* __restrict__ out)
{
    __shared__ float red[128];
    __shared__ float w1[HID];
    const int b = blockIdx.x;
    const int tid = threadIdx.x;
    if (tid < HID) w1[tid] = Wfc1[tid];
    __syncthreads();
    float s = 0.f;
    for (int t = tid; t < TT; t += 128) {
        const v8bf* hp = (const v8bf*)(h8 + ((size_t)b * TT + t) * HID);
        float d = bfc1[0];
        #pragma unroll
        for (int q = 0; q < 4; ++q) {
            v8bf hv = hp[q];
            #pragma unroll
            for (int k = 0; k < 8; ++k) d += (float)hv[k] * w1[q * 8 + k];
        }
        s += Wfc2[t] * d;
    }
    red[tid] = s;
    __syncthreads();
    for (int off = 64; off > 0; off >>= 1) {
        if (tid < off) red[tid] += red[tid + off];
        __syncthreads();
    }
    if (tid == 0) out[b] = red[0] + bfc2[0];
}

// ---------------------------------------------------------------------------
extern "C" void kernel_launch(void* const* d_in, const int* in_sizes, int n_in,
                              void* d_out, int out_size, void* d_ws, size_t ws_size,
                              hipStream_t stream)
{
    const float* x    = (const float*)d_in[0];
    const float* W0   = (const float*)d_in[1];
    const float* b0   = (const float*)d_in[2];
    const float* W1   = (const float*)d_in[3];
    const float* b1   = (const float*)d_in[4];
    const float* Wih0 = (const float*)d_in[5];
    const float* Wih  = (const float*)d_in[6];
    const float* Whh  = (const float*)d_in[7];
    const float* bih  = (const float*)d_in[8];
    const float* bhh  = (const float*)d_in[9];
    const float* Wfc1 = (const float*)d_in[10];
    const float* bfc1 = (const float*)d_in[11];
    const float* Wfc2 = (const float*)d_in[12];
    const float* bfc2 = (const float*)d_in[13];
    float* out = (float*)d_out;

    char* ws = (char*)d_ws;
    __bf16* hbuf  = (__bf16*)ws;                             // 19200*64*2  = 2457600
    __bf16* hlay  = (__bf16*)(ws + 2457600);                 // 8*19200*32*2 = 9830400
    int*    flags = (int*)(ws + 2457600 + 9830400);          // 9600 ints

    zero_flags_kernel<<<(9600 + 255) / 256, 256, 0, stream>>>(flags, 9600);
    fc0_kernel<<<150, 256, 0, stream>>>(x, W0, b0, W1, b1, hbuf);
    lstm_kernel<<<32, 256, 0, stream>>>(hbuf, Wih0, Wih, Whh, bih, bhh, hlay, flags);
    head_kernel<<<BB, 128, 0, stream>>>(hlay + (size_t)(NL - 1) * BT * HID,
                                        Wfc1, bfc1, Wfc2, bfc2, out);
}
